// MultiplexInductiveSmoother_14207751815941
// MI455X (gfx1250) — compile-verified, matrix operand order checked by reading, not run on hardware
//
#include <hip/hip_runtime.h>
#include <hip/hip_bf16.h>
#include <math.h>

typedef __attribute__((ext_vector_type(16))) _Float16 v16h;
typedef __attribute__((ext_vector_type(8)))  float    v8f;

#define BASELINE 6.0f
#define LN_EPS   1e-5f

// ---------------------------------------------------------------------------
// K0: zero edge-histogram counters and v_prior accumulator
// ---------------------------------------------------------------------------
__global__ void k_init(int* counts, int nCounts, float* v_prior, int P) {
    int g = blockIdx.x * blockDim.x + threadIdx.x;
    if (g < nCounts) counts[g] = 0;
    if (g < P)       v_prior[g] = 0.0f;
}

// ---------------------------------------------------------------------------
// K1: per-edge binary search (searchsorted) + histogram of segment ids
// ---------------------------------------------------------------------------
__global__ void k_hist(const int* ei0_f, const int* ei0_r,
                       const float* y_f, const float* y_r,
                       const int* nb_f, const int* nb_r,
                       int* counts, int* posArr, float* wArr,
                       int N, int E) {
    int g = blockIdx.x * blockDim.x + threadIdx.x;
    if (g >= 2 * E) return;
    int L = (g >= E) ? 1 : 0;
    int e = g - L * E;
    const int*   ei0 = L ? ei0_r : ei0_f;
    const float* y   = L ? y_r   : y_f;
    const int*   nb  = L ? nb_r  : nb_f;

    int v = ei0[e];
    int lo = 0, hi = N;                      // lower_bound (searchsorted left)
    while (lo < hi) {
        int mid = (lo + hi) >> 1;
        if (nb[mid] < v) lo = mid + 1; else hi = mid;
    }
    int pos = lo > N - 1 ? N - 1 : lo;       // clip like reference
    if (nb[pos] == v) {
        posArr[g] = pos;
        wArr[g]   = y[e] - BASELINE;
        atomicAdd(&counts[L * N + pos], 1);
    } else {
        posArr[g] = -1;
    }
}

// ---------------------------------------------------------------------------
// K2: exclusive scan of per-neighbor counts (one block per layer, N <= 2048)
// ---------------------------------------------------------------------------
__global__ void k_scan(const int* counts, int* offsets, int* cursors, int N) {
    __shared__ int s[2048];
    int L = blockIdx.x, t = threadIdx.x;            // 1024 threads
    s[t]        = (t        < N) ? counts[L * N + t]        : 0;
    s[t + 1024] = (t + 1024 < N) ? counts[L * N + t + 1024] : 0;
    __syncthreads();
    for (int off = 1; off < 2048; off <<= 1) {
        int a = (t >= off)        ? s[t - off]        : 0;
        int b = (t + 1024 >= off) ? s[t + 1024 - off] : 0;
        __syncthreads();
        s[t] += a; s[t + 1024] += b;
        __syncthreads();
    }
    if (t < N) {
        int ex = (t == 0) ? 0 : s[t - 1];
        offsets[L * (N + 1) + t] = ex;
        cursors[L * N + t]       = ex;
    }
    int t2 = t + 1024;
    if (t2 < N) {
        int ex = s[t2 - 1];
        offsets[L * (N + 1) + t2] = ex;
        cursors[L * N + t2]       = ex;
    }
    if (t == 0) offsets[L * (N + 1) + N] = s[N - 1];
}

// ---------------------------------------------------------------------------
// K3: scatter edge ids into segment-sorted order
// ---------------------------------------------------------------------------
__global__ void k_scatter(const int* posArr, int* cursors, int* sorted,
                          int N, int E) {
    int g = blockIdx.x * blockDim.x + threadIdx.x;
    if (g >= 2 * E) return;
    int L = (g >= E) ? 1 : 0;
    int e = g - L * E;
    int pos = posArr[g];
    if (pos >= 0) {
        int idx = atomicAdd(&cursors[L * N + pos], 1);
        sorted[(size_t)L * E + idx] = e;
    }
}

// ---------------------------------------------------------------------------
// K4: per-neighbor gather-accumulate (registers, no atomics). Dominant kernel:
//     coalesced float4 row loads, unroll-2 + prefetch for MLP.
// ---------------------------------------------------------------------------
__global__ void k_gather(const int* __restrict__ sorted,
                         const int* __restrict__ offsets,
                         const float* __restrict__ wArr,
                         const int* __restrict__ ei1_f,
                         const int* __restrict__ ei1_r,
                         const float4* __restrict__ drugs,
                         float* __restrict__ out_msgs,  // d_out + P
                         int N, int E, int D) {
    int n = blockIdx.x, L = blockIdx.y, t = threadIdx.x;  // 128 threads
    const int*   ei1 = L ? ei1_r : ei1_f;
    const int*   srt = sorted + (size_t)L * E;
    const float* w   = wArr   + (size_t)L * E;
    int start = offsets[L * (N + 1) + n];
    int end   = offsets[L * (N + 1) + n + 1];
    int vpr   = D >> 2;                                   // float4 per row

    float4 acc; acc.x = acc.y = acc.z = acc.w = 0.0f;
    int j = start;
    for (; j + 1 < end; j += 2) {
        int e0 = srt[j], e1 = srt[j + 1];
        int d0 = ei1[e0], d1 = ei1[e1];
        float w0 = w[e0], w1 = w[e1];
        if (j + 3 < end) {  // prefetch next pair's rows (global_prefetch_b8)
            __builtin_prefetch(&drugs[(size_t)ei1[srt[j + 2]] * vpr + t], 0, 0);
            __builtin_prefetch(&drugs[(size_t)ei1[srt[j + 3]] * vpr + t], 0, 0);
        }
        float4 x0 = drugs[(size_t)d0 * vpr + t];
        float4 x1 = drugs[(size_t)d1 * vpr + t];
        acc.x += w0 * x0.x + w1 * x1.x;
        acc.y += w0 * x0.y + w1 * x1.y;
        acc.z += w0 * x0.z + w1 * x1.z;
        acc.w += w0 * x0.w + w1 * x1.w;
    }
    if (j < end) {
        int e0 = srt[j]; int d0 = ei1[e0]; float w0 = w[e0];
        float4 x0 = drugs[(size_t)d0 * vpr + t];
        acc.x += w0 * x0.x; acc.y += w0 * x0.y;
        acc.z += w0 * x0.z; acc.w += w0 * x0.w;
    }
    ((float4*)out_msgs)[((size_t)L * N + n) * vpr + t] = acc;
}

// ---------------------------------------------------------------------------
// K5a: stage A fragments: features -> f16 in WMMA 16x32 fragment order.
//      One 32B fragment per (layer, mtile, kstep, lane); coalesced writes.
// ---------------------------------------------------------------------------
__global__ void k_stageA(const float* __restrict__ formF,
                         const float* __restrict__ roleF,
                         _Float16* __restrict__ zst, int N, int P) {
    int mt = blockIdx.x, L = blockIdx.y;
    int tid = threadIdx.x;                     // 512 = (P/32 ksteps) * 32 lanes
    int ks = tid >> 5, lane = tid & 31;
    const float* Z = L ? roleF : formF;
    int l15 = lane & 15, half = lane >> 4;
    int row = mt * 16 + l15;
    int mtiles = N / 16, ksteps = P / 32;
    size_t frag = ((size_t)(L * mtiles + mt) * ksteps + ks);
    _Float16* dst = zst + (frag * 32 + lane) * 16;
#pragma unroll
    for (int i = 0; i < 16; ++i) {
        int k = ((i < 8) ? i : i + 8) + half * 8;
        dst[i] = (_Float16)Z[(size_t)row * P + ks * 32 + k];
    }
}

// ---------------------------------------------------------------------------
// K5b: stage B fragments: W1[P:2P] -> f16, 32x16 fragment order per
//      (kstep, ntile, lane). 64 KB total, L2-resident for all waves.
// ---------------------------------------------------------------------------
__global__ void k_stageB(const float* __restrict__ W1,
                         _Float16* __restrict__ w1st, int P) {
    int tid = blockIdx.x * blockDim.x + threadIdx.x;   // (P/32)*4*32
    int ksteps = P / 32;
    if (tid >= ksteps * 4 * 32) return;
    int lane = tid & 31, nt = (tid >> 5) & 3, ks = tid >> 7;
    int l15 = lane & 15, half = lane >> 4;
    _Float16* dst = w1st + (size_t)tid * 16;           // frag = (ks*4+nt)*32+lane
#pragma unroll
    for (int i = 0; i < 16; ++i) {
        int k = ((i < 8) ? i : i + 8) + half * 8;
        dst[i] = (_Float16)W1[(size_t)(P + ks * 32 + k) * 64 + nt * 16 + l15];
    }
}

// ---------------------------------------------------------------------------
// K5c: constant part of attention input: target@W1_top + lemb@W1_bot + b1
// ---------------------------------------------------------------------------
__global__ void k_const0(const float* target, const float* lemb,
                         const float* W1, const float* b1,
                         float* const0, int P) {
    int t = threadIdx.x;          // 128 = 2 layers * 64 cols
    int col = t & 63, L = t >> 6;
    float s = b1[col];
    for (int p = 0; p < P; ++p)      s += target[p] * W1[(size_t)p * 64 + col];
    for (int jj = 0; jj < 16; ++jj)  s += lemb[L * 16 + jj] * W1[(size_t)(2 * P + jj) * 64 + col];
    const0[L * 64 + col] = s;
}

// ---------------------------------------------------------------------------
// K6: attention logits via WMMA. One wave per 16-row tile per layer.
//     Inner loop is pure b128 fragment loads + 4x v_wmma_f32_16x16x32_f16.
// ---------------------------------------------------------------------------
__global__ void __launch_bounds__(32)
k_attn(const _Float16* __restrict__ zst, const _Float16* __restrict__ w1st,
       const float* __restrict__ W2, const float* __restrict__ b2,
       const float* __restrict__ const0, float* __restrict__ logits,
       int N, int P) {
    int lane = threadIdx.x;               // wave32
    int mt = blockIdx.x, L = blockIdx.y;
    int half = lane >> 4;
    int l15  = lane & 15;
    int mtiles = N / 16, ksteps = P / 32;

    const v16h* av = (const v16h*)zst;
    const v16h* bv = (const v16h*)w1st;
    size_t abase = (size_t)(L * mtiles + mt) * ksteps;

    v8f acc[4] = {};                      // 4 N-tiles of 16 (64 cols)
    for (int ks = 0; ks < ksteps; ++ks) {
        v16h a = av[(abase + ks) * 32 + lane];
        v16h b0 = bv[(ks * 4 + 0) * 32 + lane];
        v16h b1 = bv[(ks * 4 + 1) * 32 + lane];
        v16h b2f = bv[(ks * 4 + 2) * 32 + lane];
        v16h b3 = bv[(ks * 4 + 3) * 32 + lane];
        acc[0] = __builtin_amdgcn_wmma_f32_16x16x32_f16(false, a, false, b0,  (short)0, acc[0], false, false);
        acc[1] = __builtin_amdgcn_wmma_f32_16x16x32_f16(false, a, false, b1,  (short)0, acc[1], false, false);
        acc[2] = __builtin_amdgcn_wmma_f32_16x16x32_f16(false, a, false, b2f, (short)0, acc[2], false, false);
        acc[3] = __builtin_amdgcn_wmma_f32_16x16x32_f16(false, a, false, b3,  (short)0, acc[3], false, false);
    }

    // Epilogue: C/D layout => VGPR r holds M = r + 8*half, lane&15 = N
    float bb2 = b2[0];
#pragma unroll
    for (int r = 0; r < 8; ++r) {
        float s = 0.0f;
#pragma unroll
        for (int nt = 0; nt < 4; ++nt) {
            float h = acc[nt][r] + const0[L * 64 + nt * 16 + l15];
            h = (h > 0.0f) ? h : 0.2f * h;               // LeakyReLU(0.2)
            s += h * W2[nt * 16 + l15];
        }
        s += __shfl_xor(s, 1, 32);
        s += __shfl_xor(s, 2, 32);
        s += __shfl_xor(s, 4, 32);
        s += __shfl_xor(s, 8, 32);                        // sum over 16 cols
        if (l15 == 0)
            logits[(size_t)L * N + mt * 16 + r + half * 8] = s + bb2;
    }
}

// ---------------------------------------------------------------------------
// K7: softmax over all 2N logits (single block)
// ---------------------------------------------------------------------------
__global__ void k_softmax(const float* logits, float* wnorm, int M) {
    __shared__ float red[1024];
    __shared__ float sMax, sSum;
    int t = threadIdx.x;
    float m = -1e30f;
    for (int i = t; i < M; i += 1024) m = fmaxf(m, logits[i]);
    red[t] = m; __syncthreads();
    for (int off = 512; off > 0; off >>= 1) {
        if (t < off) red[t] = fmaxf(red[t], red[t + off]);
        __syncthreads();
    }
    if (t == 0) sMax = red[0];
    __syncthreads();
    float s = 0.0f;
    for (int i = t; i < M; i += 1024) s += __expf(logits[i] - sMax);
    red[t] = s; __syncthreads();
    for (int off = 512; off > 0; off >>= 1) {
        if (t < off) red[t] += red[t + off];
        __syncthreads();
    }
    if (t == 0) sSum = red[0];
    __syncthreads();
    float inv = 1.0f / sSum;
    for (int i = t; i < M; i += 1024)
        wnorm[i] = __expf(logits[i] - sMax) * inv;
}

// ---------------------------------------------------------------------------
// K8: v_prior = sum_i wnorm[i] * msgs[i]   (msgs live in d_out, L2-resident)
// ---------------------------------------------------------------------------
__global__ void k_vprior(const float* __restrict__ wnorm,
                         const float* __restrict__ msgs,
                         float* v_prior, int M, int D, int rowsPer) {
    int t = threadIdx.x;                 // D threads
    int r0 = blockIdx.x * rowsPer;
    float acc = 0.0f;
    for (int i = 0; i < rowsPer; ++i) {
        int r = r0 + i;
        if (r < M) acc += wnorm[r] * msgs[(size_t)r * D + t];
    }
    atomicAdd(&v_prior[t], acc);
}

// ---------------------------------------------------------------------------
// K9: h = PReLU(v_prior @ Wi1 + bi1)
// ---------------------------------------------------------------------------
__global__ void k_mlp1(const float* v_prior, const float* Wi1, const float* bi1,
                       const float* alpha, float* hbuf, int P) {
    int j = threadIdx.x;                 // P threads, coalesced over Wi1 rows
    float s = bi1[j];
    for (int d = 0; d < P; ++d) s += v_prior[d] * Wi1[(size_t)d * P + j];
    float a = alpha[0];
    hbuf[j] = (s > 0.0f) ? s : a * s;
}

// ---------------------------------------------------------------------------
// K10: v = h @ Wi2 + bi2; x = target + v; LayerNorm -> d_out[0:P]
// ---------------------------------------------------------------------------
__global__ void k_mlp2_ln(const float* hbuf, const float* Wi2, const float* bi2,
                          const float* target, const float* gamma,
                          const float* beta, float* out, int P) {
    __shared__ float red[512];
    __shared__ float sMu, sVar;
    int j = threadIdx.x;
    float s = bi2[j];
    for (int k = 0; k < P; ++k) s += hbuf[k] * Wi2[(size_t)k * P + j];
    float x = target[j] + s;
    red[j] = x; __syncthreads();
    for (int off = 256; off > 0; off >>= 1) {
        if (j < off) red[j] += red[j + off];
        __syncthreads();
    }
    if (j == 0) sMu = red[0] / (float)P;
    __syncthreads();
    float d = x - sMu;
    red[j] = d * d; __syncthreads();
    for (int off = 256; off > 0; off >>= 1) {
        if (j < off) red[j] += red[j + off];
        __syncthreads();
    }
    if (j == 0) sVar = red[0] / (float)P;
    __syncthreads();
    out[j] = d * rsqrtf(sVar + LN_EPS) * gamma[j] + beta[j];
}

// ---------------------------------------------------------------------------
extern "C" void kernel_launch(void* const* d_in, const int* in_sizes, int n_in,
                              void* d_out, int out_size, void* d_ws, size_t ws_size,
                              hipStream_t stream) {
    const float* target = (const float*)d_in[0];
    const float* formF  = (const float*)d_in[1];
    const float* roleF  = (const float*)d_in[2];
    const int*   nb_f   = (const int*)d_in[3];
    const int*   nb_r   = (const int*)d_in[4];
    const int*   ei_f   = (const int*)d_in[5];
    const float* y_f    = (const float*)d_in[6];
    const int*   ei_r   = (const int*)d_in[7];
    const float* y_r    = (const float*)d_in[8];
    const float* drugs  = (const float*)d_in[9];
    const float* lemb   = (const float*)d_in[10];
    const float* W1     = (const float*)d_in[11];
    const float* b1     = (const float*)d_in[12];
    const float* W2     = (const float*)d_in[13];
    const float* b2     = (const float*)d_in[14];
    const float* Wi1    = (const float*)d_in[15];
    const float* bi1    = (const float*)d_in[16];
    const float* alpha  = (const float*)d_in[17];
    const float* Wi2    = (const float*)d_in[18];
    const float* bi2    = (const float*)d_in[19];
    const float* gamma  = (const float*)d_in[20];
    const float* beta   = (const float*)d_in[21];

    const int P = in_sizes[0];        // 512
    const int N = in_sizes[3];        // 2000
    const int E = in_sizes[6];        // 300000
    const int D = P;
    const int ksteps = P / 32;

    // ---- workspace carve-up (256B aligned) ----
    size_t off = 0;
    auto carve = [&](size_t bytes) -> char* {
        off = (off + 255) & ~(size_t)255;
        char* p = (char*)d_ws + off;
        off += bytes;
        return p;
    };
    int*   counts  = (int*)  carve(sizeof(int)   * 2 * N);
    int*   offsets = (int*)  carve(sizeof(int)   * 2 * (N + 1));
    int*   cursors = (int*)  carve(sizeof(int)   * 2 * N);
    int*   posArr  = (int*)  carve(sizeof(int)   * 2 * (size_t)E);
    int*   sorted  = (int*)  carve(sizeof(int)   * 2 * (size_t)E);
    float* wArr    = (float*)carve(sizeof(float) * 2 * (size_t)E);
    _Float16* zst  = (_Float16*)carve(sizeof(_Float16) * 2 * (size_t)N * P);
    _Float16* w1st = (_Float16*)carve(sizeof(_Float16) * (size_t)ksteps * 4 * 32 * 16);
    float* const0  = (float*)carve(sizeof(float) * 128);
    float* logits  = (float*)carve(sizeof(float) * 2 * N);
    float* wnorm   = (float*)carve(sizeof(float) * 2 * N);
    float* v_prior = (float*)carve(sizeof(float) * P);
    float* hbuf    = (float*)carve(sizeof(float) * P);

    float* out      = (float*)d_out;
    float* out_msgs = out + P;         // form_msgs then role_msgs, contiguous

    // ---- pref_vectors path (counting-sort segment sum) ----
    int initN = 2 * N > P ? 2 * N : P;
    k_init<<<(initN + 255) / 256, 256, 0, stream>>>(counts, 2 * N, v_prior, P);

    int edgeBlocks = (2 * E + 255) / 256;
    k_hist<<<edgeBlocks, 256, 0, stream>>>(ei_f, ei_r, y_f, y_r, nb_f, nb_r,
                                           counts, posArr, wArr, N, E);
    k_scan<<<2, 1024, 0, stream>>>(counts, offsets, cursors, N);
    k_scatter<<<edgeBlocks, 256, 0, stream>>>(posArr, cursors, sorted, N, E);
    k_gather<<<dim3(N, 2), 128, 0, stream>>>(sorted, offsets, wArr,
                                             ei_f + E, ei_r + E,
                                             (const float4*)drugs, out_msgs,
                                             N, E, D);

    // ---- attention path (WMMA with staged f16 fragments) ----
    k_stageA<<<dim3(N / 16, 2), ksteps * 32, 0, stream>>>(formF, roleF, zst, N, P);
    k_stageB<<<(ksteps * 4 * 32 + 255) / 256, 256, 0, stream>>>(W1, w1st, P);
    k_const0<<<1, 128, 0, stream>>>(target, lemb, W1, b1, const0, P);
    k_attn<<<dim3(N / 16, 2), 32, 0, stream>>>(zst, w1st, W2, b2,
                                               const0, logits, N, P);
    k_softmax<<<1, 1024, 0, stream>>>(logits, wnorm, 2 * N);

    int rowsPer = (2 * N + 31) / 32;
    k_vprior<<<32, D, 0, stream>>>(wnorm, out_msgs, v_prior, 2 * N, D, rowsPer);

    // ---- interaction MLP + layernorm ----
    k_mlp1<<<1, P, 0, stream>>>(v_prior, Wi1, bi1, alpha, hbuf, P);
    k_mlp2_ln<<<1, P, 0, stream>>>(hbuf, Wi2, bi2, target, gamma, beta, out, P);
}
